// RoutedNetwork_68135361183805
// MI455X (gfx1250) — matrix operand order repaired and checked by reading, not compiled, not run on hardware
//
#include <hip/hip_runtime.h>

// ---- sizes (fixed by the reference) ----
#define B_  32
#define S_  50
#define L_  20
#define R_  16
#define V_  50000
#define D_  100
#define T_  320          // L*R tokens per sentence
#define RD_ 1600         // R*D
#define M_  1600         // S*B sentence instances

typedef __attribute__((ext_vector_type(16))) _Float16 v16h;
typedef __attribute__((ext_vector_type(8)))  _Float16 v8h;
typedef __attribute__((ext_vector_type(8)))  float    v8f;

// A 16x32 f16 fragment load: lane holds K = half*16+{0..15} of row m
__device__ __forceinline__ v16h load_a_frag(const _Float16* __restrict__ row, int k0, int half) {
  v8h lo = *(const v8h*)(row + k0 + half * 16);
  v8h hi = *(const v8h*)(row + k0 + half * 16 + 8);
  v16h a;
#pragma unroll
  for (int t = 0; t < 8; ++t) { a[t] = lo[t]; a[t + 8] = hi[t]; }
  return a;
}

// ---------------- prep kernels ----------------

// v_keys[i] = sum_j keys_flat[j] * Vw[i][j]   (one block per output, LDS tree reduce)
__global__ void vk_kernel(const float* __restrict__ Vw, const float* __restrict__ keys,
                          float* __restrict__ vk) {
  __shared__ float red[256];
  int i = blockIdx.x;
  float s = 0.f;
  for (int j = threadIdx.x; j < RD_; j += 256) s += Vw[i * RD_ + j] * keys[j];
  red[threadIdx.x] = s;
  __syncthreads();
  for (int w = 128; w > 0; w >>= 1) {
    if (threadIdx.x < w) red[threadIdx.x] += red[threadIdx.x + w];
    __syncthreads();
  }
  if (threadIdx.x == 0) vk[i] = red[0];
}

// Ut[k][rd] = U[rd][k]  (transpose so scan reads are coalesced along rd)
__global__ void ut_kernel(const float* __restrict__ U, float* __restrict__ Ut) {
  int rd = blockIdx.x;
  int k = threadIdx.x;
  if (k < D_) Ut[k * RD_ + rd] = U[rd * D_ + k];
}

// generic f32 -> f16 convert
__global__ void h_convert(const float* __restrict__ src, _Float16* __restrict__ dst, int n) {
  int i = blockIdx.x * blockDim.x + threadIdx.x;
  if (i < n) dst[i] = (_Float16)src[i];
}

// Rw [V,100] f32 -> Rwh [V,128] f16, zero-padded K (so K-loop of 4x32 is exact)
__global__ void rw_convert(const float* __restrict__ Rw, _Float16* __restrict__ Rwh) {
  int i = blockIdx.x * blockDim.x + threadIdx.x;
  if (i >= V_ * 128) return;
  int v = i >> 7;
  int c = i & 127;
  Rwh[i] = (c < D_) ? (_Float16)Rw[v * D_ + c] : (_Float16)0.f;
}

// ---------------- gathers ----------------

// X[(s*B+b)*1600 + r*100 + d] = sum_{j<20} emb0[stories[b][s][r*20+j]][d] * in_mult[r*20+j][d]
__global__ void story_gather(const int* __restrict__ stories, const float* __restrict__ emb,
                             const float* __restrict__ in_mult,
                             float* __restrict__ X, _Float16* __restrict__ Xh) {
  int blk = blockIdx.x;                 // s*(B*R) + b*R + r
  int r = blk % R_;
  int b = (blk / R_) % B_;
  int s = blk / (R_ * B_);
  int d = threadIdx.x;
  if (d >= D_) return;
  const int* srow = stories + (b * S_ + s) * T_ + r * 20;
  float acc = 0.f;
#pragma unroll 4
  for (int j = 0; j < 20; ++j) {
    int tok = srow[j];
    float e = (tok == 0) ? 0.f : emb[tok * D_ + d];   // padding_idx=0 row is zero
    acc += e * in_mult[(r * 20 + j) * D_ + d];
  }
  int idx = (s * B_ + b) * RD_ + r * D_ + d;
  X[idx] = acc;
  Xh[idx] = (_Float16)acc;
}

// Q[b][d] = sum_l emb0[queries[b][l]][d] * q_mult[l][d]
__global__ void query_gather(const int* __restrict__ queries, const float* __restrict__ emb,
                             const float* __restrict__ q_mult, float* __restrict__ Q) {
  int b = blockIdx.x;
  int d = threadIdx.x;
  if (d >= D_) return;
  float acc = 0.f;
  for (int l = 0; l < L_; ++l) {
    int tok = queries[b * L_ + l];
    float e = (tok == 0) ? 0.f : emb[tok * D_ + d];
    acc += e * q_mult[l * D_ + d];
  }
  Q[b * D_ + d] = acc;
}

// ---------------- big WMMA GEMM: WV[m][n] = sum_k Xh[m][k]*Ww[n][k] + vk[n] ----------------
// M=N=K=1600, f16 in / f32 acc. Each wave: 32x64 output tile (2 A frags x 4 B frags
// = 8 WMMAs per 32-deep k-step). k-loop unrolled x2 with ping-pong fragment buffers:
// each buffer is loaded once and consumed in place -> prefetch overlaps WMMAs with
// no register-rotation moves.
__global__ void __launch_bounds__(256)
gemm_ww(const _Float16* __restrict__ Xh, const _Float16* __restrict__ Wh,
        const float* __restrict__ vk, float* __restrict__ WV) {
  const int K = RD_, N = RD_;
  int job = (blockIdx.x * blockDim.x + threadIdx.x) >> 5;   // uniform per wave
  if (job >= (M_ / 32) * (N / 64)) return;                  // 50*25 = 1250
  int mpair = job / (N / 64);
  int ngrp  = job % (N / 64);
  int lane  = threadIdx.x & 31;
  int half  = lane >> 4;          // 0: lanes 0-15, 1: lanes 16-31
  int l16   = lane & 15;

  v8f c[2][4];
#pragma unroll
  for (int j = 0; j < 4; ++j) {
    float v = vk[ngrp * 64 + j * 16 + l16];   // C init = v_keys[n], same for all rows
#pragma unroll
    for (int t = 0; t < 8; ++t) { c[0][j][t] = v; c[1][j][t] = v; }
  }

  const _Float16* arow0 = Xh + (mpair * 32 + l16) * K;
  const _Float16* arow1 = arow0 + 16 * K;
  const _Float16* brow  = Wh + (ngrp * 64 + l16) * K;   // B tile j at +j*16*K

  // ping-pong fragment buffers
  v16h a0_0, a1_0, b_0[4];   // buffer 0
  v16h a0_1, a1_1, b_1[4];   // buffer 1

#define LOAD_STEP(k0, A0, A1, Bv)                                              \
  do {                                                                         \
    A0 = load_a_frag(arow0, (k0), half);                                       \
    A1 = load_a_frag(arow1, (k0), half);                                       \
    _Pragma("unroll") for (int j = 0; j < 4; ++j)                              \
        Bv[j] = *(const v16h*)(brow + j * 16 * K + (k0) + half * 16);          \
  } while (0)

#define WMMA_STEP(A0, A1, Bv)                                                  \
  do {                                                                         \
    _Pragma("unroll") for (int j = 0; j < 4; ++j) {                            \
      c[0][j] = __builtin_amdgcn_wmma_f32_16x16x32_f16(false, A0, false, Bv[j],\
                                                       (short)0, c[0][j],      \
                                                       false, false);          \
      c[1][j] = __builtin_amdgcn_wmma_f32_16x16x32_f16(false, A1, false, Bv[j],\
                                                       (short)0, c[1][j],      \
                                                       false, false);          \
    }                                                                          \
  } while (0)

  LOAD_STEP(0, a0_0, a1_0, b_0);
  for (int k0 = 0; k0 < K - 64; k0 += 64) {
    LOAD_STEP(k0 + 32, a0_1, a1_1, b_1);   // prefetch odd step
    WMMA_STEP(a0_0, a1_0, b_0);            // compute even step
    LOAD_STEP(k0 + 64, a0_0, a1_0, b_0);   // prefetch next even step
    WMMA_STEP(a0_1, a1_1, b_1);            // compute odd step
  }
  // buf0 holds step K-64; load final step K-32, then drain both
  LOAD_STEP(K - 32, a0_1, a1_1, b_1);
  WMMA_STEP(a0_0, a1_0, b_0);
  WMMA_STEP(a0_1, a1_1, b_1);

#undef LOAD_STEP
#undef WMMA_STEP

  // D f32 16x16: VGPR v -> row v + 8*half, col l16
#pragma unroll
  for (int mt = 0; mt < 2; ++mt) {
    int mbase = mpair * 32 + mt * 16 + half * 8;
#pragma unroll
    for (int j = 0; j < 4; ++j) {
      int ncol = ngrp * 64 + j * 16 + l16;
#pragma unroll
      for (int v = 0; v < 8; ++v) WV[(mbase + v) * N + ncol] = c[mt][j][v];
    }
  }
}

// ---------------- recurrent scan: one block per batch element, 50 sequential steps ----------------
__global__ void __launch_bounds__(512)
scan_kernel(const float* __restrict__ X, const float* __restrict__ WV,
            const float* __restrict__ Ut, const float* __restrict__ keys,
            const float* __restrict__ init_state, const float* __restrict__ alpha_in,
            float* __restrict__ STATE) {
  __shared__ float st[D_];
  __shared__ float contrib[RD_];
  int b = blockIdx.x;
  float alpha = alpha_in[0];
  for (int d = threadIdx.x; d < D_; d += blockDim.x) st[d] = init_state[d];
  __syncthreads();
  for (int s = 0; s < S_; ++s) {
    const float* wvrow = WV + (s * B_ + b) * RD_;
    const float* xrow  = X  + (s * B_ + b) * RD_;
    for (int rd = threadIdx.x; rd < RD_; rd += blockDim.x) {
      int d = rd % D_;
      float u = 0.f;
#pragma unroll 4
      for (int k = 0; k < D_; ++k) u += st[k] * Ut[k * RD_ + rd];   // u = state @ U.T
      float pre  = u + wvrow[rd];                                   // + v_keys + w
      float cand = (pre >= 0.f) ? pre : alpha * pre;                // PReLU
      float sent = xrow[rd];
      float g    = sent * (st[d] + keys[rd]);
      float gate = 1.f / (1.f + expf(-g));
      float msk  = (gate == 0.5f) ? 0.f : 1.f;                      // null-sentence mask
      contrib[rd] = cand * gate * msk;
    }
    __syncthreads();
    for (int d = threadIdx.x; d < D_; d += blockDim.x) {
      float v = st[d];
      for (int r = 0; r < R_; ++r) v += contrib[r * D_ + d];        // deterministic order
      st[d] = v / sqrtf(v * v);                                     // faithful sign-normalize
    }
    __syncthreads();
  }
  for (int d = threadIdx.x; d < D_; d += blockDim.x) STATE[b * D_ + d] = st[d];
}

// ---------------- h = prelu(q + state @ Hw.T) -> f16 padded [B,128] ----------------
__global__ void final_h(const float* __restrict__ STATE, const float* __restrict__ Q,
                        const float* __restrict__ Hw, const float* __restrict__ alpha_q,
                        _Float16* __restrict__ Hh) {
  int b = blockIdx.x;
  int d = threadIdx.x;   // 128 threads
  if (d < D_) {
    float acc = 0.f;
#pragma unroll 4
    for (int k = 0; k < D_; ++k) acc += STATE[b * D_ + k] * Hw[d * D_ + k];
    float h = Q[b * D_ + d] + acc;
    float a = alpha_q[0];
    h = (h >= 0.f) ? h : a * h;
    Hh[b * 128 + d] = (_Float16)h;
  } else if (d < 128) {
    Hh[b * 128 + d] = (_Float16)0.f;
  }
}

// ---------------- out[b][v] = sum_d h[b][d]*Rw[v][d]  (M=32, N=50000, K=128) ----------------
// K is only 4 WMMA steps: issue all 8 fragment loads as one clause, then 4 WMMAs.
__global__ void __launch_bounds__(256)
gemm_out(const _Float16* __restrict__ Hh, const _Float16* __restrict__ Rwh,
         float* __restrict__ out) {
  const int K = 128, N = V_;
  int job = (blockIdx.x * blockDim.x + threadIdx.x) >> 5;   // uniform per wave
  if (job >= 2 * (N / 16)) return;                          // 2*3125 = 6250
  int mtile = job / (N / 16);
  int ntile = job % (N / 16);
  int lane = threadIdx.x & 31;
  int half = lane >> 4;
  int l16  = lane & 15;

  const _Float16* arow = Hh + (mtile * 16 + l16) * K;
  const _Float16* brow = Rwh + (ntile * 16 + l16) * K;

  v16h a[4], b[4];
#pragma unroll
  for (int s = 0; s < 4; ++s) {
    a[s] = load_a_frag(arow, s * 32, half);
    b[s] = *(const v16h*)(brow + s * 32 + half * 16);
  }

  v8f c;
#pragma unroll
  for (int t = 0; t < 8; ++t) c[t] = 0.f;
#pragma unroll
  for (int s = 0; s < 4; ++s)
    c = __builtin_amdgcn_wmma_f32_16x16x32_f16(false, a[s], false, b[s], (short)0, c,
                                               false, false);

  int mbase = mtile * 16 + half * 8;
#pragma unroll
  for (int v = 0; v < 8; ++v) out[(mbase + v) * N + ntile * 16 + l16] = c[v];
}

// ---------------- launcher ----------------
extern "C" void kernel_launch(void* const* d_in, const int* in_sizes, int n_in,
                              void* d_out, int out_size, void* d_ws, size_t ws_size,
                              hipStream_t stream) {
  (void)in_sizes; (void)n_in; (void)out_size; (void)ws_size;
  const int*   stories    = (const int*)d_in[0];
  const int*   queries    = (const int*)d_in[1];
  const float* emb        = (const float*)d_in[2];
  const float* in_mult    = (const float*)d_in[3];
  const float* q_mult     = (const float*)d_in[4];
  const float* keys       = (const float*)d_in[5];
  const float* init_state = (const float*)d_in[6];
  const float* U          = (const float*)d_in[7];
  const float* Vw         = (const float*)d_in[8];
  const float* Ww         = (const float*)d_in[9];
  const float* Hw         = (const float*)d_in[10];
  const float* Rw         = (const float*)d_in[11];
  const float* alpha_in   = (const float*)d_in[12];
  const float* alpha_q    = (const float*)d_in[13];
  float* out = (float*)d_out;

  // workspace layout (all offsets 256B aligned)
  char* ws = (char*)d_ws;
  float*    vk  = (float*)(ws + 0);           //   1600 f32
  float*    Ut  = (float*)(ws + 6400);        // 160000 f32
  float*    X   = (float*)(ws + 646400);      // 1600*1600 f32
  float*    WV  = (float*)(ws + 10886400);    // 1600*1600 f32
  _Float16* Xh  = (_Float16*)(ws + 21126400); // 1600*1600 f16
  _Float16* Wh  = (_Float16*)(ws + 26246400); // 1600*1600 f16
  _Float16* Rwh = (_Float16*)(ws + 31366400); // 50000*128 f16
  float*    Q   = (float*)(ws + 44166400);    // 32*100 f32
  float*    ST  = (float*)(ws + 44179200);    // 32*100 f32
  _Float16* Hh  = (_Float16*)(ws + 44192000); // 32*128 f16

  vk_kernel   <<<RD_,   256, 0, stream>>>(Vw, keys, vk);
  ut_kernel   <<<RD_,   128, 0, stream>>>(U, Ut);
  h_convert   <<<10000, 256, 0, stream>>>(Ww, Wh, M_ * RD_);
  rw_convert  <<<25000, 256, 0, stream>>>(Rw, Rwh);
  story_gather<<<S_ * B_ * R_, 128, 0, stream>>>(stories, emb, in_mult, X, Xh);
  query_gather<<<B_,   128, 0, stream>>>(queries, emb, q_mult, Q);
  gemm_ww     <<<157,  256, 0, stream>>>(Xh, Wh, vk, WV);     // 1250 wave-jobs
  scan_kernel <<<B_,   512, 0, stream>>>(X, WV, Ut, keys, init_state, alpha_in, ST);
  final_h     <<<B_,   128, 0, stream>>>(ST, Q, Hw, alpha_q, Hh);
  gemm_out    <<<782,  256, 0, stream>>>(Hh, Rwh, out);       // 6250 wave-jobs
}
